// CausalSelfAttention_17669495456006
// MI455X (gfx1250) — compile-verified, matrix operand order checked by reading, not run on hardware
//
#include <hip/hip_runtime.h>
#include <hip/hip_bf16.h>

typedef __attribute__((ext_vector_type(16))) __bf16 v16bf;
typedef __attribute__((ext_vector_type(8)))  __bf16 v8bf;
typedef __attribute__((ext_vector_type(4)))  __bf16 v4bf;
typedef __attribute__((ext_vector_type(8)))  float  v8f;
typedef __attribute__((ext_vector_type(4)))  int    v4i;

#define WMMA_BF16(a, b, c) \
  __builtin_amdgcn_wmma_f32_16x16x32_bf16(false, (a), false, (b), (short)0, (c), false, false)

#if defined(__HIP_DEVICE_COMPILE__) && __has_builtin(__builtin_amdgcn_global_load_async_to_lds_b128)
#define HAS_ASYNC_LDS 1
#else
#define HAS_ASYNC_LDS 0
#endif

#if HAS_ASYNC_LDS
__device__ __forceinline__ void async_copy16(const __bf16* g, __bf16* l) {
  __builtin_amdgcn_global_load_async_to_lds_b128(
      (__attribute__((address_space(1))) v4i*)g,
      (__attribute__((address_space(3))) v4i*)l, 0, 0);
}
__device__ __forceinline__ void wait_async0() {
#if __has_builtin(__builtin_amdgcn_s_wait_asynccnt)
  __builtin_amdgcn_s_wait_asynccnt(0);
#else
  asm volatile("s_wait_asynccnt 0x0" ::: "memory");
#endif
}
#endif

__device__ __forceinline__ v16bf cat8(v8bf a, v8bf b) {
  return __builtin_shufflevector(a, b, 0,1,2,3,4,5,6,7,8,9,10,11,12,13,14,15);
}
__device__ __forceinline__ v8bf ld8(const __bf16* p) { return *(const v8bf*)p; }

// ---------------------------------------------------------------------------
// fp32 -> bf16 conversion (x4 vectorized)
// ---------------------------------------------------------------------------
__global__ void cvt_kernel(const float* __restrict__ in, __bf16* __restrict__ out, int n4) {
  int i = blockIdx.x * blockDim.x + threadIdx.x;
  if (i >= n4) return;
  float4 f = ((const float4*)in)[i];
  v4bf o = { (__bf16)f.x, (__bf16)f.y, (__bf16)f.z, (__bf16)f.w };
  ((v4bf*)out)[i] = o;
}

// ---------------------------------------------------------------------------
// Tiled transpose (+ optional fp32->bf16 convert): dst[c][r] = src[r][c]
// 64x64 tiles via padded LDS. z-batched: src += (z/zdiv)*srcZ1 + (z%zdiv)*srcZ0,
// dst += z*dstZ0. One-shot memory-bound prep; negligible vs compute.
// ---------------------------------------------------------------------------
template <typename TIN>
__global__ __launch_bounds__(256) void transpose_to_bf16(
    const TIN* __restrict__ src, __bf16* __restrict__ dst,
    int srcRS, int dstRS, int zdiv, long srcZ1, long srcZ0, long dstZ0) {
  __shared__ __bf16 tile[64][72];
  const int z = blockIdx.z;
  src += (size_t)(z / zdiv) * srcZ1 + (size_t)(z % zdiv) * srcZ0;
  dst += (size_t)z * dstZ0;
  const int r0 = blockIdx.y * 64, c0 = blockIdx.x * 64;
  const int t = threadIdx.x;
  const int lr = t >> 2, lc = (t & 3) * 16;

  const TIN* sp = src + (size_t)(r0 + lr) * srcRS + c0 + lc;
#pragma unroll
  for (int i = 0; i < 16; i++) tile[lr][lc + i] = (__bf16)sp[i];
  __syncthreads();

  __bf16* dp = dst + (size_t)(c0 + lr) * dstRS + r0 + lc;
  v8bf o0, o1;
#pragma unroll
  for (int i = 0; i < 8; i++) { o0[i] = tile[lc + i][lr]; o1[i] = tile[lc + 8 + i][lr]; }
  *(v8bf*)dp       = o0;
  *(v8bf*)(dp + 8) = o1;
}

// ---------------------------------------------------------------------------
// bf16 WMMA GEMM: C[M,N] = A[M,K] * BT[N,K]^T + bias   (both row-major, bf16)
// block tile 128x128, 8 waves (4x2), wave tile 32x64, k-chunk 32.
// Double-buffered LDS; tiles staged with GLOBAL_LOAD_ASYNC_TO_LDS_B128 so the
// chunk-(k+1) copy overlaps the chunk-k WMMA burst (ASYNCcnt pipeline).
// ---------------------------------------------------------------------------
template <bool OUT_BF16>
__global__ __launch_bounds__(256) void gemm_bf16(
    const __bf16* __restrict__ A, const __bf16* __restrict__ BT,
    const float* __restrict__ bias, void* __restrict__ Cout,
    int M, int N, int K) {
  constexpr int LDT = 40;          // padded row stride (halves)
  constexpr int BUF = 128 * LDT;   // one tile buffer
  __shared__ __bf16 sA[2 * BUF];   // A tile  [m][k], double buffered
  __shared__ __bf16 sB[2 * BUF];   // BT tile [n][k], double buffered

  const int tid  = threadIdx.x;
  const int lane = tid & 31;
  const int wave = tid >> 5;
  const int half = lane >> 4;
  const int lm   = lane & 15;
  const int wm   = (wave & 3) * 32;
  const int wn   = (wave >> 2) * 64;
  const int mBase = blockIdx.y * 128;
  const int nBase = blockIdx.x * 128;

  v8f acc[2][4];
#pragma unroll
  for (int i = 0; i < 2; i++)
#pragma unroll
    for (int j = 0; j < 4; j++) acc[i][j] = {};

  const int srow = tid >> 1;        // 0..127
  const int scol = (tid & 1) * 16;  // 0 / 16
  const __bf16* gA = A  + (size_t)(mBase + srow) * K + scol;
  const __bf16* gB = BT + (size_t)(nBase + srow) * K + scol;
  __bf16* lA = &sA[srow * LDT + scol];
  __bf16* lB = &sB[srow * LDT + scol];

#if HAS_ASYNC_LDS
  // prologue: chunk 0 -> buffer 0
  async_copy16(gA, lA);
  async_copy16(gA + 8, lA + 8);
  async_copy16(gB, lB);
  async_copy16(gB + 8, lB + 8);
#endif

  int p = 0;
  for (int kb = 0; kb < K; kb += 32, p ^= 1) {
#if HAS_ASYNC_LDS
    wait_async0();       // my chunk-kb copies have landed in LDS
    __syncthreads();     // everyone's landed; all waves done reading buf p^1
    if (kb + 32 < K) {   // kick chunk kb+32 into the other buffer
      async_copy16(gA + kb + 32, lA + (p ^ 1) * BUF);
      async_copy16(gA + kb + 40, lA + (p ^ 1) * BUF + 8);
      async_copy16(gB + kb + 32, lB + (p ^ 1) * BUF);
      async_copy16(gB + kb + 40, lB + (p ^ 1) * BUF + 8);
    }
#else
    if (kb + 32 < K) {   // prefetch next chunk (global_prefetch_b8)
      __builtin_prefetch(gA + kb + 32, 0, 1);
      __builtin_prefetch(gB + kb + 32, 0, 1);
    }
    __syncthreads();
    *(v8bf*)(lA + p * BUF)     = ld8(gA + kb);
    *(v8bf*)(lA + p * BUF + 8) = ld8(gA + kb + 8);
    *(v8bf*)(lB + p * BUF)     = ld8(gB + kb);
    *(v8bf*)(lB + p * BUF + 8) = ld8(gB + kb + 8);
    __syncthreads();
#endif

    const __bf16* bufA = &sA[p * BUF];
    const __bf16* bufB = &sB[p * BUF];
    v16bf af[2], bfr[4];
#pragma unroll
    for (int i = 0; i < 2; i++) {
      int r = wm + i * 16 + lm;
      af[i] = cat8(ld8(&bufA[r * LDT + 8 * half]), ld8(&bufA[r * LDT + 16 + 8 * half]));
    }
#pragma unroll
    for (int j = 0; j < 4; j++) {
      int n = wn + j * 16 + lm;
      bfr[j] = cat8(ld8(&bufB[n * LDT + 8 * half]), ld8(&bufB[n * LDT + 16 + 8 * half]));
    }
#pragma unroll
    for (int i = 0; i < 2; i++)
#pragma unroll
      for (int j = 0; j < 4; j++) acc[i][j] = WMMA_BF16(af[i], bfr[j], acc[i][j]);
  }

  // epilogue: bias + store (C layout: lane half picks m-block, vgpr r = m%8)
#pragma unroll
  for (int i = 0; i < 2; i++) {
#pragma unroll
    for (int j = 0; j < 4; j++) {
      int n    = nBase + wn + j * 16 + lm;
      float bv = bias[n];
#pragma unroll
      for (int r = 0; r < 8; r++) {
        int m   = mBase + wm + i * 16 + half * 8 + r;
        float v = acc[i][j][r] + bv;
        if (OUT_BF16)
          ((__bf16*)Cout)[(size_t)m * N + n] = (__bf16)v;
        else
          ((float*)Cout)[(size_t)m * N + n] = v;
      }
    }
  }
}

// ---------------------------------------------------------------------------
// Flash attention. Block = 64 q-rows x (b,h); 4 waves, 16 q-rows/wave.
// K frags: rows of K (= columns of K^T) straight from global.
// V frags: rows of pre-transposed V^T straight from global.
// Only per-wave P relayout (C->A) goes through LDS; no block barriers at all.
// ---------------------------------------------------------------------------
__global__ __launch_bounds__(128) void attn_kernel(
    const __bf16* __restrict__ qkv, const __bf16* __restrict__ vT,
    __bf16* __restrict__ attout) {
  constexpr int SEQ = 2048, HD = 128, E3 = 6144, E = 2048, LDT = 40;
  __shared__ __bf16 sP[4 * 16 * LDT];  // per-wave P tile [16][32]

  const int tid  = threadIdx.x;
  const int lane = tid & 31, wave = tid >> 5;
  const int half = lane >> 4, lm = lane & 15;
  const int qt = blockIdx.x;  // 64-row query tile
  const int bh = blockIdx.y;
  const int b = bh >> 4, h = bh & 15;

  const __bf16* Qb  = qkv + (size_t)b * SEQ * E3 + h * HD;
  const __bf16* Kb  = Qb + E;
  const __bf16* vTb = vT + (size_t)bh * HD * SEQ;  // [d][s]

  // Q fragments: 4 k-chunks of 32, kept in registers for the whole key loop
  const int qrow_lane = qt * 64 + wave * 16 + lm;
  v16bf qf[4];
  {
    const __bf16* qp = Qb + (size_t)qrow_lane * E3;
#pragma unroll
    for (int c = 0; c < 4; c++)
      qf[c] = cat8(ld8(qp + c * 32 + 8 * half), ld8(qp + c * 32 + 16 + 8 * half));
  }

  v8f oacc[8];
#pragma unroll
  for (int j = 0; j < 8; j++) oacc[j] = {};
  float mi[8], li[8];
#pragma unroll
  for (int r = 0; r < 8; r++) { mi[r] = -__builtin_inff(); li[r] = 0.f; }

  const float scale = 0.08838834764831845f;  // 1/sqrt(128)
  const int kend = qt * 64 + 64;             // causal bound for this tile
  const int pbase = wave * 16 * LDT;

  for (int ks = 0; ks < kend; ks += 32) {
    if (ks + 32 < kend) {  // prefetch next chunk streams
      __builtin_prefetch(Kb + (size_t)(ks + 32 + lm) * E3, 0, 1);
      __builtin_prefetch(vTb + (size_t)(lm * 16) * SEQ + ks + 32, 0, 1);
    }

    // S = Q K^T (two 16-key subtiles)
    v8f sa[2];
    sa[0] = {}; sa[1] = {};
#pragma unroll
    for (int t = 0; t < 2; t++) {
      const __bf16* kp = Kb + (size_t)(ks + t * 16 + lm) * E3;
#pragma unroll
      for (int c = 0; c < 4; c++) {
        v16bf kf = cat8(ld8(kp + c * 32 + 8 * half), ld8(kp + c * 32 + 16 + 8 * half));
        sa[t] = WMMA_BF16(qf[c], kf, sa[t]);
      }
    }

    // causal mask + online softmax in C-matrix layout
    float sc[2][8], rmax[8];
#pragma unroll
    for (int r = 0; r < 8; r++) rmax[r] = -__builtin_inff();
#pragma unroll
    for (int t = 0; t < 2; t++) {
      int keyn = ks + t * 16 + lm;
#pragma unroll
      for (int r = 0; r < 8; r++) {
        int qr  = qt * 64 + wave * 16 + half * 8 + r;
        float v = sa[t][r] * scale;
        if (keyn > qr) v = -__builtin_inff();
        sc[t][r] = v;
        rmax[r]  = fmaxf(rmax[r], v);
      }
    }
#pragma unroll
    for (int msk = 1; msk < 16; msk <<= 1)
#pragma unroll
      for (int r = 0; r < 8; r++) rmax[r] = fmaxf(rmax[r], __shfl_xor(rmax[r], msk, 32));

#pragma unroll
    for (int r = 0; r < 8; r++) {
      float mnew  = fmaxf(mi[r], rmax[r]);
      float alpha = __expf(mi[r] - mnew);
      float rs = 0.f;
#pragma unroll
      for (int t = 0; t < 2; t++) {
        float p = __expf(sc[t][r] - mnew);
        rs += p;
        sP[pbase + (half * 8 + r) * LDT + t * 16 + lm] = (__bf16)p;
      }
#pragma unroll
      for (int msk = 1; msk < 16; msk <<= 1) rs += __shfl_xor(rs, msk, 32);
      li[r] = li[r] * alpha + rs;
      mi[r] = mnew;
#pragma unroll
      for (int j = 0; j < 8; j++) oacc[j][r] *= alpha;
    }

    // O += P V : A = P (re-layout via per-wave LDS), B = rows of V^T (global)
    v16bf pf = cat8(ld8(&sP[pbase + lm * LDT + 8 * half]),
                    ld8(&sP[pbase + lm * LDT + 16 + 8 * half]));
#pragma unroll
    for (int j = 0; j < 8; j++) {
      const __bf16* vp = vTb + (size_t)(j * 16 + lm) * SEQ + ks;
      v16bf vf = cat8(ld8(vp + 8 * half), ld8(vp + 16 + 8 * half));
      oacc[j] = WMMA_BF16(pf, vf, oacc[j]);
    }
  }

  // epilogue: normalize by row-sum; heads recombined into [b, s, h*128+d]
#pragma unroll
  for (int j = 0; j < 8; j++) {
#pragma unroll
    for (int r = 0; r < 8; r++) {
      int qr  = qt * 64 + wave * 16 + half * 8 + r;
      int col = h * HD + j * 16 + lm;
      attout[(size_t)(b * SEQ + qr) * E + col] = (__bf16)(oacc[j][r] / li[r]);
    }
  }
}

// ---------------------------------------------------------------------------
extern "C" void kernel_launch(void* const* d_in, const int* in_sizes, int n_in,
                              void* d_out, int out_size, void* d_ws, size_t ws_size,
                              hipStream_t stream) {
  const float* x    = (const float*)d_in[0];
  const float* Wqkv = (const float*)d_in[1];
  const float* bqkv = (const float*)d_in[2];
  const float* Wout = (const float*)d_in[3];
  const float* bout = (const float*)d_in[4];

  __bf16* xb   = (__bf16*)d_ws;                  // 4096*2048
  __bf16* wqT  = xb   + (size_t)4096 * 2048;     // 6144*2048  (W_qkv^T)
  __bf16* woT  = wqT  + (size_t)6144 * 2048;     // 2048*2048  (W_out^T)
  __bf16* qkvb = woT  + (size_t)2048 * 2048;     // 4096*6144  (Q|K|V, bias fused)
  __bf16* vTw  = qkvb + (size_t)4096 * 6144;     // 32*128*2048 (V^T per (b,h))
  __bf16* attb = vTw  + (size_t)32 * 128 * 2048; // 4096*2048

  // x -> bf16
  {
    int n4 = (int)((size_t)4096 * 2048 / 4);
    cvt_kernel<<<(n4 + 255) / 256, 256, 0, stream>>>(x, xb, n4);
  }
  // W_qkv^T, W_out^T: fused fp32->bf16 convert + transpose
  transpose_to_bf16<float><<<dim3(6144 / 64, 2048 / 64, 1), 256, 0, stream>>>(
      Wqkv, wqT, 6144, 2048, 1, 0, 0, 0);
  transpose_to_bf16<float><<<dim3(2048 / 64, 2048 / 64, 1), 256, 0, stream>>>(
      Wout, woT, 2048, 2048, 1, 0, 0, 0);

  // qkv = x @ W_qkv + b_qkv  -> bf16
  gemm_bf16<true><<<dim3(6144 / 128, 4096 / 128), 256, 0, stream>>>(
      xb, wqT, bqkv, qkvb, 4096, 6144, 2048);

  // V^T per (b,h): [2048 s x 128 d] -> [128 d x 2048 s]
  transpose_to_bf16<__bf16><<<dim3(128 / 64, 2048 / 64, 32), 256, 0, stream>>>(
      qkvb + 4096, vTw, 6144, 2048, 16, (long)2048 * 6144, 128, (long)128 * 2048);

  // causal flash attention -> bf16 [b, s, e]
  attn_kernel<<<dim3(32, 32), 128, 0, stream>>>(qkvb, vTw, attb);

  // out = attn @ W_out + b_out -> fp32
  gemm_bf16<false><<<dim3(2048 / 128, 4096 / 128), 256, 0, stream>>>(
      attb, woT, bout, d_out, 4096, 2048, 2048);
}
